// ResUNet_W1A1_44186623541990
// MI455X (gfx1250) — 8-bit matrix operand order repaired and checked by reading, compiled, not run on hardware
//
#include <hip/hip_runtime.h>
#include <math.h>

#define TIME_DIM 32
#define BN_EPS 1e-5f

typedef __attribute__((ext_vector_type(8))) int   v8i_t;
typedef __attribute__((ext_vector_type(2))) float v2f_t;
typedef __attribute__((ext_vector_type(8))) float v8f_t;

__device__ __forceinline__ float gelu_exact(float x) {
    return 0.5f * x * (1.0f + erff(x * 0.70710678118654752f));
}

__global__ void k_time_emb(const float* __restrict__ t,
                           const float* __restrict__ w,
                           const float* __restrict__ bias,
                           float* __restrict__ out, int B) {
    __shared__ float emb[16][TIME_DIM];
    int tid = threadIdx.x;
    int b = tid / TIME_DIM, j = tid % TIME_DIM;
    if (b < B) {
        const int half = TIME_DIM / 2;
        float val;
        if (j < half) {
            float f = expf((float)j * (-logf(10000.0f) / (float)(half - 1)));
            val = sinf(t[b] * f);
        } else {
            int jj = j - half;
            float f = expf((float)jj * (-logf(10000.0f) / (float)(half - 1)));
            val = cosf(t[b] * f);
        }
        emb[b][j] = val;
    }
    __syncthreads();
    if (b < B) {
        float acc = bias[j];
        for (int k = 0; k < TIME_DIM; ++k) acc += emb[b][k] * w[j * TIME_DIM + k];
        out[b * TIME_DIM + j] = gelu_exact(acc);
    }
}

__global__ void k_temb_proj(const float* __restrict__ temb, const float* __restrict__ w,
                            const float* __restrict__ bias, float* __restrict__ out,
                            int B, int O) {
    int i = blockIdx.x * blockDim.x + threadIdx.x;
    if (i >= B * O) return;
    int b = i / O, o = i % O;
    float acc = bias[o];
    for (int k = 0; k < TIME_DIM; ++k) acc += temb[b * TIME_DIM + k] * w[o * TIME_DIM + k];
    out[i] = acc;
}

__global__ void k_conv0(const float* __restrict__ x, const float* __restrict__ w,
                        const float* __restrict__ bias, float* __restrict__ out,
                        int B, int H, int W, int O) {
    long long i = (long long)blockIdx.x * blockDim.x + threadIdx.x;
    long long total = (long long)B * O * H * W;
    if (i >= total) return;
    int xw = (int)(i % W); long long tt = i / W;
    int y = (int)(tt % H); tt /= H;
    int o = (int)(tt % O); int b = (int)(tt / O);
    float acc = bias[o];
    for (int dy = 0; dy < 3; ++dy) {
        int yy = y + dy - 1; if (yy < 0 || yy >= H) continue;
        for (int dx = 0; dx < 3; ++dx) {
            int xx = xw + dx - 1; if (xx < 0 || xx >= W) continue;
            acc += x[((long long)b * H + yy) * W + xx] * w[o * 9 + dy * 3 + dx];
        }
    }
    out[i] = acc;
}

__global__ void k_maxpool(const float* __restrict__ x, float* __restrict__ out,
                          int B, int C, int H, int W) {
    int Ho = H / 2, Wo = W / 2;
    long long i = (long long)blockIdx.x * blockDim.x + threadIdx.x;
    long long total = (long long)B * C * Ho * Wo;
    if (i >= total) return;
    int xo = (int)(i % Wo); long long t = i / Wo;
    int yo = (int)(t % Ho); t /= Ho;
    int c = (int)(t % C); int b = (int)(t / C);
    const float* p = x + (((long long)b * C + c) * H + yo * 2) * W + xo * 2;
    out[i] = fmaxf(fmaxf(p[0], p[1]), fmaxf(p[W], p[W + 1]));
}

__global__ void k_bn_sign_nhwc(const float* __restrict__ srcA, int Ca, int upA,
                               const float* __restrict__ srcB, int Cb,
                               const float* __restrict__ g, const float* __restrict__ bb,
                               const float* __restrict__ mean, const float* __restrict__ var,
                               signed char* __restrict__ act,
                               int B, int H, int W) {
    int C = Ca + Cb;
    long long i = (long long)blockIdx.x * blockDim.x + threadIdx.x;
    long long total = (long long)B * H * W * C;
    if (i >= total) return;
    int c = (int)(i % C); long long t = i / C;
    int x = (int)(t % W); t /= W;
    int y = (int)(t % H); int b = (int)(t / H);
    float v;
    if (c < Ca) {
        int ys = upA ? (y >> 1) : y, xs = upA ? (x >> 1) : x;
        int Hs = upA ? (H >> 1) : H, Ws = upA ? (W >> 1) : W;
        v = srcA[(((long long)b * Ca + c) * Hs + ys) * Ws + xs];
    } else {
        v = srcB[(((long long)b * Cb + (c - Ca)) * H + y) * W + x];
    }
    float inv = g[c] * rsqrtf(var[c] + BN_EPS);
    float h = v * inv + (bb[c] - mean[c] * inv);
    act[i] = (h >= 0.0f) ? (signed char)1 : (signed char)-1;
}

__global__ void k_prep_w3(const float* __restrict__ w, signed char* __restrict__ Bt,
                          float* __restrict__ alpha, int O, int C) {
    int o = blockIdx.x;
    int K = C * 9;
    __shared__ float red[256];
    float s = 0.0f;
    for (int k = threadIdx.x; k < K; k += blockDim.x) {
        int c = k / 9, tap = k % 9;
        float wv = w[((long long)o * C + c) * 9 + tap];
        s += fabsf(wv);
        Bt[(long long)o * K + tap * C + c] = (wv >= 0.0f) ? (signed char)1 : (signed char)-1;
    }
    red[threadIdx.x] = s; __syncthreads();
    for (int st = 128; st > 0; st >>= 1) {
        if ((int)threadIdx.x < st) red[threadIdx.x] += red[threadIdx.x + st];
        __syncthreads();
    }
    if (threadIdx.x == 0) alpha[o] = red[0] / (float)K;
}

__global__ void k_prep_w1T(const float* __restrict__ w, float* __restrict__ weffT,
                           int O, int C) {
    int o = blockIdx.x;
    __shared__ float red[256];
    float s = 0.0f;
    for (int c = threadIdx.x; c < C; c += blockDim.x) s += fabsf(w[(long long)o * C + c]);
    red[threadIdx.x] = s; __syncthreads();
    for (int st = 128; st > 0; st >>= 1) {
        if ((int)threadIdx.x < st) red[threadIdx.x] += red[threadIdx.x + st];
        __syncthreads();
    }
    float a = red[0] / (float)C;
    __syncthreads();
    for (int c = threadIdx.x; c < C; c += blockDim.x)
        weffT[(long long)c * O + o] = (w[(long long)o * C + c] >= 0.0f) ? a : -a;
}

__global__ void __launch_bounds__(256)
k_bitconv_wmma(const signed char* __restrict__ act,
               const signed char* __restrict__ Bt,
               const float* __restrict__ alpha,
               const float* __restrict__ temb,
               float* __restrict__ out,
               int B, int H, int W, int C, int O) {
    const int K    = 9 * C;
    const int lane = threadIdx.x & 31;
    const int wave = threadIdx.x >> 5;
    const int half = lane >> 4;
    const int ncol = lane & 15;
    const int n0   = blockIdx.y * 64;
    const int m0   = (blockIdx.x * 8 + wave) * 16;
    const int HW   = H * W;

    const int mrow = m0 + (lane & 15);
    const int ax = mrow % W;
    const int ay = (mrow / W) % H;
    const int ab = mrow / HW;

    __shared__ __align__(16) signed char ldsB[64 * 64];

    v8i_t acc[4];
#pragma unroll
    for (int j = 0; j < 4; ++j) acc[j] = (v8i_t)0;

    for (int kb = 0; kb < K; kb += 64) {
        __syncthreads();
        {
            int nl  = threadIdx.x >> 2;
            int seg = threadIdx.x & 3;
            const signed char* src = Bt + (long long)(n0 + nl) * K + kb + seg * 16;
            unsigned loff = (unsigned)(size_t)&ldsB[nl * 64 + seg * 16];
            asm volatile("global_load_async_to_lds_b128 %0, %1, off"
                         :: "v"(loff), "v"(src) : "memory");
        }
        asm volatile("s_wait_asynccnt 0" ::: "memory");
        __syncthreads();

        int tap = kb / C;
        int c0  = kb - tap * C;
        int dy = tap / 3 - 1, dx = tap % 3 - 1;
        int yy = ay + dy, xx = ax + dx;
        union { v8i_t v; unsigned long long q[4]; } A;
        if (yy >= 0 && yy < H && xx >= 0 && xx < W) {
            const unsigned long long* p = (const unsigned long long*)
                (act + ((((long long)ab * H + yy) * W + xx) * C + c0 + half * 16));
            A.q[0] = p[0];
            A.q[1] = p[1];
            A.q[2] = p[4];
            A.q[3] = p[5];
        } else {
            A.q[0] = A.q[1] = A.q[2] = A.q[3] = 0ull;
        }

#pragma unroll
        for (int j = 0; j < 4; ++j) {
            const int col = j * 16 + ncol;
            union { v8i_t v; int4 h2[2]; } Bf;
            Bf.h2[0] = *(const int4*)&ldsB[col * 64 + half * 16];
            Bf.h2[1] = *(const int4*)&ldsB[col * 64 + 32 + half * 16];
            acc[j] = __builtin_amdgcn_wmma_i32_16x16x64_iu8(
                          true, A.v,  true, Bf.v, acc[j],
                          false,  false);
        }
    }

    const int brow = m0 / HW;
#pragma unroll
    for (int j = 0; j < 4; ++j) {
        const int n = n0 + j * 16 + ncol;
        const float al = alpha[n];
        const float tadd = temb ? temb[brow * O + n] : 0.0f;
#pragma unroll
        for (int i = 0; i < 8; ++i) {
            int mr = m0 + 8 * half + i;
            int yx = mr % HW;
            int oy = yx / W, ox = yx % W;
            out[((((long long)brow * O) + n) * H + oy) * W + ox] =
                (float)acc[j][i] * al + tadd;
        }
    }
}

__device__ __forceinline__ float load_cat(const float* __restrict__ srcA, int Ca, int upA,
                                          const float* __restrict__ srcB, int Cb,
                                          int b, int y, int x, int H, int W, int c) {
    if (c < Ca) {
        int ys = upA ? (y >> 1) : y, xs = upA ? (x >> 1) : x;
        int Hs = upA ? (H >> 1) : H, Ws = upA ? (W >> 1) : W;
        return srcA[(((long long)b * Ca + c) * Hs + ys) * Ws + xs];
    }
    return srcB[(((long long)b * Cb + (c - Ca)) * H + y) * W + x];
}

__global__ void __launch_bounds__(256)
k_skip_wmma(const float* __restrict__ srcA, int Ca, int upA,
            const float* __restrict__ srcB, int Cb,
            const float* __restrict__ weffT,
            float* __restrict__ out,
            int B, int H, int W, int O) {
    const int C    = Ca + Cb;
    const int lane = threadIdx.x & 31;
    const int wave = threadIdx.x >> 5;
    const int half = lane >> 4;
    const int ncol = lane & 15;
    const int n0   = blockIdx.y * 16;
    const int m0   = (blockIdx.x * 8 + wave) * 16;
    const int HW   = H * W;

    const int mrow = m0 + (lane & 15);
    const int ax = mrow % W;
    const int ay = (mrow / W) % H;
    const int ab = mrow / HW;
    const int brow = m0 / HW;

    v8f_t acc;
#pragma unroll
    for (int i = 0; i < 8; ++i) {
        int mr = m0 + 8 * half + i;
        int yx = mr % HW;
        int oy = yx / W, ox = yx % W;
        acc[i] = out[((((long long)brow * O) + (n0 + ncol)) * H + oy) * W + ox];
    }

    for (int k0 = 0; k0 < C; k0 += 4) {
        int c = k0 + 2 * half;
        v2f_t a, bv;
        a[0]  = load_cat(srcA, Ca, upA, srcB, Cb, ab, ay, ax, H, W, c);
        a[1]  = load_cat(srcA, Ca, upA, srcB, Cb, ab, ay, ax, H, W, c + 1);
        bv[0] = weffT[(long long)c * O + n0 + ncol];
        bv[1] = weffT[(long long)(c + 1) * O + n0 + ncol];
        acc = __builtin_amdgcn_wmma_f32_16x16x4_f32(
                   false, a,  false, bv,
                   (short)0, acc,  false,  false);
    }

#pragma unroll
    for (int i = 0; i < 8; ++i) {
        int mr = m0 + 8 * half + i;
        int yx = mr % HW;
        int oy = yx / W, ox = yx % W;
        out[((((long long)brow * O) + (n0 + ncol)) * H + oy) * W + ox] = acc[i];
    }
}

__global__ void k_out_conv(const float* __restrict__ u2, const float* __restrict__ w,
                           const float* __restrict__ bias, float* __restrict__ out,
                           int B, int C, int H, int W) {
    long long i = (long long)blockIdx.x * blockDim.x + threadIdx.x;
    long long total = (long long)B * H * W;
    if (i >= total) return;
    int x = (int)(i % W); long long t = i / W;
    int y = (int)(t % H); int b = (int)(t / H);
    float acc = bias[0];
    for (int c = 0; c < C; ++c)
        acc += u2[(((long long)b * C + c) * H + y) * W + x] * w[c];
    out[i] = acc;
}

extern "C" void kernel_launch(void* const* d_in, const int* in_sizes, int n_in,
                              void* d_out, int out_size, void* d_ws, size_t ws_size,
                              hipStream_t stream) {
    (void)in_sizes; (void)n_in; (void)out_size; (void)ws_size;
    const int B = 16;

    const float* x       = (const float*)d_in[0];
    const float* t       = (const float*)d_in[1];
    const float* tmlp_w  = (const float*)d_in[2];
    const float* tmlp_b  = (const float*)d_in[3];
    const float* conv0_w = (const float*)d_in[4];
    const float* conv0_b = (const float*)d_in[5];
    const int DOWN1 = 6, DOWN2 = 19, UP1 = 32, UP2 = 45;
    const float* out_w   = (const float*)d_in[58];
    const float* out_b   = (const float*)d_in[59];

    char* ws = (char*)d_ws;
    size_t off = 0;
    auto alloc = [&](size_t bytes) -> char* {
        char* p = ws + off;
        off = (off + bytes + 255) & ~(size_t)255;
        return p;
    };
    float* TEMB  = (float*)alloc((size_t)B * TIME_DIM * 4);
    float* TPD1  = (float*)alloc((size_t)B * 128 * 4);
    float* TPD2  = (float*)alloc((size_t)B * 256 * 4);
    float* TPU1  = (float*)alloc((size_t)B * 128 * 4);
    float* TPU2  = (float*)alloc((size_t)B * 64 * 4);
    float* X0    = (float*)alloc((size_t)B * 64 * 128 * 128 * 4);
    float* P1    = (float*)alloc((size_t)B * 64 * 64 * 64 * 4);
    float* X1    = (float*)alloc((size_t)B * 128 * 64 * 64 * 4);
    float* P2    = (float*)alloc((size_t)B * 128 * 32 * 32 * 4);
    float* X2    = (float*)alloc((size_t)B * 256 * 32 * 32 * 4);
    float* U1    = (float*)alloc((size_t)B * 128 * 64 * 64 * 4);
    float* U2    = (float*)alloc((size_t)B * 64 * 128 * 128 * 4);
    float* HBUF  = (float*)alloc((size_t)B * 64 * 128 * 128 * 4);
    signed char* ACT = (signed char*)alloc((size_t)B * 128 * 128 * 192);
    signed char* BT  = (signed char*)alloc((size_t)256 * 9 * 256);
    float* ALPHA = (float*)alloc(256 * 4);
    float* WEFFT = (float*)alloc((size_t)384 * 128 * 4);

    auto cdiv = [](long long a, long long b) { return (unsigned)((a + b - 1) / b); };

    k_time_emb<<<1, 512, 0, stream>>>(t, tmlp_w, tmlp_b, TEMB, B);

    k_conv0<<<cdiv((long long)B * 64 * 128 * 128, 256), 256, 0, stream>>>(
        x, conv0_w, conv0_b, X0, B, 128, 128, 64);
    k_maxpool<<<cdiv((long long)B * 64 * 64 * 64, 256), 256, 0, stream>>>(
        X0, P1, B, 64, 128, 128);

    auto run_block = [&](int p, const float* srcA, int Ca, int upA,
                         const float* srcB, int Cb,
                         int H, int W, int O, float* tproj, float* outbuf) {
        const int C = Ca + Cb;
        const float* tm_w   = (const float*)d_in[p + 0];
        const float* tm_b   = (const float*)d_in[p + 1];
        const float* bn1_g  = (const float*)d_in[p + 2];
        const float* bn1_b  = (const float*)d_in[p + 3];
        const float* bn1_m  = (const float*)d_in[p + 4];
        const float* bn1_v  = (const float*)d_in[p + 5];
        const float* w1     = (const float*)d_in[p + 6];
        const float* bn2_g  = (const float*)d_in[p + 7];
        const float* bn2_b  = (const float*)d_in[p + 8];
        const float* bn2_m  = (const float*)d_in[p + 9];
        const float* bn2_v  = (const float*)d_in[p + 10];
        const float* w2     = (const float*)d_in[p + 11];
        const float* skip_w = (const float*)d_in[p + 12];
        const long long M = (long long)B * H * W;

        k_temb_proj<<<cdiv((long long)B * O, 256), 256, 0, stream>>>(
            TEMB, tm_w, tm_b, tproj, B, O);

        k_bn_sign_nhwc<<<cdiv(M * C, 256), 256, 0, stream>>>(
            srcA, Ca, upA, srcB, Cb, bn1_g, bn1_b, bn1_m, bn1_v, ACT, B, H, W);
        k_prep_w3<<<O, 256, 0, stream>>>(w1, BT, ALPHA, O, C);
        {
            dim3 grid((unsigned)(M / 128), (unsigned)(O / 64));
            k_bitconv_wmma<<<grid, 256, 0, stream>>>(ACT, BT, ALPHA, tproj, HBUF,
                                                     B, H, W, C, O);
        }
        k_bn_sign_nhwc<<<cdiv(M * O, 256), 256, 0, stream>>>(
            HBUF, O, 0, nullptr, 0, bn2_g, bn2_b, bn2_m, bn2_v, ACT, B, H, W);
        k_prep_w3<<<O, 256, 0, stream>>>(w2, BT, ALPHA, O, O);
        {
            dim3 grid((unsigned)(M / 128), (unsigned)(O / 64));
            k_bitconv_wmma<<<grid, 256, 0, stream>>>(ACT, BT, ALPHA, nullptr, outbuf,
                                                     B, H, W, O, O);
        }
        k_prep_w1T<<<O, 256, 0, stream>>>(skip_w, WEFFT, O, C);
        {
            dim3 grid((unsigned)(M / 128), (unsigned)(O / 16));
            k_skip_wmma<<<grid, 256, 0, stream>>>(srcA, Ca, upA, srcB, Cb, WEFFT,
                                                  outbuf, B, H, W, O);
        }
    };

    run_block(DOWN1, P1, 64, 0, nullptr, 0, 64, 64, 128, TPD1, X1);
    k_maxpool<<<cdiv((long long)B * 128 * 32 * 32, 256), 256, 0, stream>>>(
        X1, P2, B, 128, 64, 64);
    run_block(DOWN2, P2, 128, 0, nullptr, 0, 32, 32, 256, TPD2, X2);

    run_block(UP1, X2, 256, 1, X1, 128, 64, 64, 128, TPU1, U1);
    run_block(UP2, U1, 128, 1, X0, 64, 128, 128, 64, TPU2, U2);

    k_out_conv<<<cdiv((long long)B * 128 * 128, 256), 256, 0, stream>>>(
        U2, out_w, out_b, (float*)d_out, B, 64, 128, 128);
}
